// GRUModel_85727547228527
// MI455X (gfx1250) — matrix operand order repaired and checked by reading, not compiled, not run on hardware
//
#include <hip/hip_runtime.h>

// GRU (3 stacked cells, H=64) over B=128, T=512, D=512.
// Phase 1: big WMMA f16 GEMM gi0 = x @ W0ih^T + b0ih for all timesteps.
// Phase 2: single persistent workgroup runs the sequential recurrence with
//          f16 WMMA (16x16x32) on LDS-resident hidden state; gi0[t] slice is
//          prefetched (global_prefetch_b8) while the layer-0 GEMM runs.
// Workspace requirement: ~50.7 MB.

#define T_STEPS 512
#define BATCH   128
#define DIN     512
#define HID     64
#define G3      192          // 3*HID
#define GSTRIDE 200          // padded LDS row stride (floats) for gate buffers

typedef __attribute__((ext_vector_type(16))) _Float16 v16h;
typedef __attribute__((ext_vector_type(8)))  _Float16 v8h;
typedef __attribute__((ext_vector_type(8)))  float    v8f;

static __device__ __forceinline__ v16h cat8(v8h lo, v8h hi) {
  return __builtin_shufflevector(lo, hi, 0,1,2,3,4,5,6,7,8,9,10,11,12,13,14,15);
}

// A-fragment (16x32 f16): lane holds 16 contiguous K = base+16*(lane>>4) .. +15 of row (lane&15)
static __device__ __forceinline__ v16h lds_afrag(const _Float16* row, int kb) {
  v8h lo = *(const v8h*)(row + kb + (kb & 8));
  v8h hi = *(const v8h*)(row + kb + (kb & 8) + 8);
  return cat8(lo, hi);
}

static __device__ __forceinline__ v8f wmma16(v16h a, v16h b, v8f c) {
  return __builtin_amdgcn_wmma_f32_16x16x32_f16(false, a, false, b, (short)0, c,
                                                false, false);
}

static __device__ __forceinline__ float sigm(float x) {
  x = fminf(fmaxf(x, -30.f), 30.f);
  return 1.0f / (1.0f + __expf(-x));
}

static __device__ __forceinline__ float tanh_f(float x) {
  x = fminf(fmaxf(x, -15.f), 15.f);
  float e = __expf(-2.0f * x);
  return (1.0f - e) / (1.0f + e);
}

// ---------------------------------------------------------------- prep: f32->f16 weights
__global__ void gru_prep_k(const float* __restrict__ W0ih, const float* __restrict__ W0hh,
                           const float* __restrict__ W1ih, const float* __restrict__ W1hh,
                           const float* __restrict__ W2ih, const float* __restrict__ W2hh,
                           _Float16* __restrict__ w0ih_h, _Float16* __restrict__ w0hh_h,
                           _Float16* __restrict__ w1ih_h, _Float16* __restrict__ w1hh_h,
                           _Float16* __restrict__ w2ih_h, _Float16* __restrict__ w2hh_h) {
  int i = blockIdx.x * blockDim.x + threadIdx.x;
  int stride = gridDim.x * blockDim.x;
  for (; i < G3 * DIN; i += stride) {
    w0ih_h[i] = (_Float16)W0ih[i];
    if (i < G3 * HID) {
      w0hh_h[i] = (_Float16)W0hh[i];
      w1ih_h[i] = (_Float16)W1ih[i];
      w1hh_h[i] = (_Float16)W1hh[i];
      w2ih_h[i] = (_Float16)W2ih[i];
      w2hh_h[i] = (_Float16)W2hh[i];
    }
  }
}

// ---------------------------------------------------------------- flags[t] = any(x[:,t,:] != 0)
__global__ void gru_flags_k(const float* __restrict__ x, int* __restrict__ flags) {
  __shared__ int anyv;
  const int t = blockIdx.x;
  if (threadIdx.x == 0) anyv = 0;
  __syncthreads();
  int found = 0;
  for (int i = threadIdx.x; i < BATCH * DIN; i += blockDim.x) {
    int b = i >> 9, d = i & (DIN - 1);
    if (x[(size_t)b * T_STEPS * DIN + (size_t)t * DIN + d] != 0.0f) { found = 1; break; }
  }
  if (found) anyv = 1;
  __syncthreads();
  if (threadIdx.x == 0) flags[t] = anyv;
}

// ---------------------------------------------------------------- gi0 = x @ W0ih^T + b0ih
// M = B*T = 65536 (8 waves/block, 1 m-tile/wave), N = 192 (12 tiles/wave), K = 512
__global__ __launch_bounds__(256) void gru_gemm_x_k(const float* __restrict__ x,
                                                    const _Float16* __restrict__ w0ih_h,
                                                    const float* __restrict__ b0ih,
                                                    float* __restrict__ gi0) {
  const int tid = threadIdx.x, lane = tid & 31, wave = tid >> 5;
  const int r15 = lane & 15, hi = lane >> 4;
  const int mt = blockIdx.x * 8 + wave;
  const int m0 = mt * 16;
  const float* arow = x + (size_t)(m0 + r15) * DIN;

  v8f acc[12] = {};
  for (int k0 = 0; k0 < DIN; k0 += 32) {
    const int kb = k0 + hi * 16;
    v8f f0 = *(const v8f*)(arow + kb);
    v8f f1 = *(const v8f*)(arow + kb + 8);
    v16h a;
#pragma unroll
    for (int i = 0; i < 8; ++i) { a[i] = (_Float16)f0[i]; a[8 + i] = (_Float16)f1[i]; }
#pragma unroll
    for (int n = 0; n < 12; ++n) {
      // B-fragment: lane holds 16 contiguous K values of row (n*16 + lane&15)
      const _Float16* brow = w0ih_h + (size_t)(n * 16 + r15) * DIN + k0 + hi * 16;
      v16h b = *(const v16h*)brow;
      acc[n] = wmma16(a, b, acc[n]);
    }
  }
#pragma unroll
  for (int n = 0; n < 12; ++n) {
    float bv = b0ih[n * 16 + r15];
#pragma unroll
    for (int r = 0; r < 8; ++r) {
      int m = m0 + r + hi * 8;               // m = b*T + t
      int t = m & (T_STEPS - 1), b = m >> 9;
      gi0[(size_t)t * (BATCH * G3) + (size_t)b * G3 + n * 16 + r15] = acc[n][r] + bv;
    }
  }
}

// ---------------------------------------------------------------- sequential recurrence
__global__ __launch_bounds__(384) void gru_rec_k(
    const float* __restrict__ gi0, const int* __restrict__ flags,
    const _Float16* __restrict__ w0hh, const _Float16* __restrict__ w1ih,
    const _Float16* __restrict__ w1hh, const _Float16* __restrict__ w2ih,
    const _Float16* __restrict__ w2hh,
    const float* __restrict__ b0hh, const float* __restrict__ b1ih,
    const float* __restrict__ b1hh, const float* __restrict__ b2ih,
    const float* __restrict__ b2hh, float* __restrict__ out) {
  __shared__ __align__(16) _Float16 h0s[BATCH * HID];
  __shared__ __align__(16) _Float16 h1s[BATCH * HID];
  __shared__ __align__(16) _Float16 h2s[BATCH * HID];
  __shared__ float GIs[BATCH * GSTRIDE];   // input-path gates (layers 1,2)
  __shared__ float GHs[BATCH * GSTRIDE];   // hidden-path gates (all layers)

  const int tid = threadIdx.x, lane = tid & 31, wave = tid >> 5;  // 12 waves
  const int r15 = lane & 15, hi = lane >> 4;
  const int n0 = wave * 16;
  const int col = n0 + r15;                 // gate column owned by this lane

  for (int i = tid; i < BATCH * HID; i += 384) {
    h0s[i] = (_Float16)0.f; h1s[i] = (_Float16)0.f; h2s[i] = (_Float16)0.f;
  }

  // Hoist B-fragments (weights are [192][64] f16 row-major; L2-resident).
  const int kbB = hi * 16;
  const size_t wrow = (size_t)col * HID;
  v16h B0h0 = *(const v16h*)(w0hh + wrow + kbB);
  v16h B0h1 = *(const v16h*)(w0hh + wrow + 32 + kbB);
  v16h B1i0 = *(const v16h*)(w1ih + wrow + kbB);
  v16h B1i1 = *(const v16h*)(w1ih + wrow + 32 + kbB);
  v16h B1h0 = *(const v16h*)(w1hh + wrow + kbB);
  v16h B1h1 = *(const v16h*)(w1hh + wrow + 32 + kbB);
  v16h B2i0 = *(const v16h*)(w2ih + wrow + kbB);
  v16h B2i1 = *(const v16h*)(w2ih + wrow + 32 + kbB);
  v16h B2h0 = *(const v16h*)(w2hh + wrow + kbB);
  v16h B2h1 = *(const v16h*)(w2hh + wrow + 32 + kbB);
  const float bv0h = b0hh[col];
  const float bv1i = b1ih[col], bv1h = b1hh[col];
  const float bv2i = b2ih[col], bv2h = b2hh[col];
  __syncthreads();

  for (int t = 0; t < T_STEPS; ++t) {
    if (flags[t] == 0) {           // inactive step: keep state, emit zeros
      for (int i = tid; i < BATCH * HID; i += 384)
        out[(size_t)(i >> 6) * (T_STEPS * HID) + (size_t)t * HID + (i & 63)] = 0.f;
      continue;
    }
    const float* git = gi0 + (size_t)t * (BATCH * G3);

    // Prefetch this step's gi0 slice (98 KB, L2-resident) toward the WGP while
    // the layer-0 WMMA GEMM runs; ew0 consumes it. Emits global_prefetch_b8.
    for (int i = tid * 32; i < BATCH * G3; i += 384 * 32)
      __builtin_prefetch(git + i, 0, 1);

    // ---- layer 0: GH = h0 @ W0hh^T + b0hh
    for (int mt = 0; mt < 8; ++mt) {
      const int m0 = mt * 16;
      const _Float16* ar = h0s + (m0 + r15) * HID;
      v16h a0 = lds_afrag(ar, hi * 8);
      v16h a1 = lds_afrag(ar, 32 + hi * 8);
      v8f c;
#pragma unroll
      for (int r = 0; r < 8; ++r) c[r] = bv0h;
      c = wmma16(a0, B0h0, c);
      c = wmma16(a1, B0h1, c);
      const int rb = m0 + hi * 8;
#pragma unroll
      for (int r = 0; r < 8; ++r) GHs[(rb + r) * GSTRIDE + col] = c[r];
    }
    __syncthreads();
    // ---- ew0: h0' = (1-z)*n + z*h0   (GI read straight from gi0[t])
    for (int i = tid; i < BATCH * HID; i += 384) {
      const int b = i >> 6, j = i & 63;
      const float* gb = git + (size_t)b * G3;
      const float* hb = GHs + b * GSTRIDE;
      float r = sigm(gb[j] + hb[j]);
      float z = sigm(gb[64 + j] + hb[64 + j]);
      float n = tanh_f(gb[128 + j] + r * hb[128 + j]);
      float hp = (float)h0s[i];
      h0s[i] = (_Float16)((1.f - z) * n + z * hp);
    }
    __syncthreads();

    // ---- layer 1: GI = h0' @ W1ih^T + b1ih ; GH = h1 @ W1hh^T + b1hh
    for (int mt = 0; mt < 8; ++mt) {
      const int m0 = mt * 16;
      const _Float16* arI = h0s + (m0 + r15) * HID;
      const _Float16* arH = h1s + (m0 + r15) * HID;
      v16h aI0 = lds_afrag(arI, hi * 8), aI1 = lds_afrag(arI, 32 + hi * 8);
      v16h aH0 = lds_afrag(arH, hi * 8), aH1 = lds_afrag(arH, 32 + hi * 8);
      v8f ci, ch;
#pragma unroll
      for (int r = 0; r < 8; ++r) { ci[r] = bv1i; ch[r] = bv1h; }
      ci = wmma16(aI0, B1i0, ci); ci = wmma16(aI1, B1i1, ci);
      ch = wmma16(aH0, B1h0, ch); ch = wmma16(aH1, B1h1, ch);
      const int rb = m0 + hi * 8;
#pragma unroll
      for (int r = 0; r < 8; ++r) {
        GIs[(rb + r) * GSTRIDE + col] = ci[r];
        GHs[(rb + r) * GSTRIDE + col] = ch[r];
      }
    }
    __syncthreads();
    for (int i = tid; i < BATCH * HID; i += 384) {
      const int b = i >> 6, j = i & 63;
      const float* ib = GIs + b * GSTRIDE;
      const float* hb = GHs + b * GSTRIDE;
      float r = sigm(ib[j] + hb[j]);
      float z = sigm(ib[64 + j] + hb[64 + j]);
      float n = tanh_f(ib[128 + j] + r * hb[128 + j]);
      float hp = (float)h1s[i];
      h1s[i] = (_Float16)((1.f - z) * n + z * hp);
    }
    __syncthreads();

    // ---- layer 2: GI = h1' @ W2ih^T + b2ih ; GH = h2 @ W2hh^T + b2hh
    for (int mt = 0; mt < 8; ++mt) {
      const int m0 = mt * 16;
      const _Float16* arI = h1s + (m0 + r15) * HID;
      const _Float16* arH = h2s + (m0 + r15) * HID;
      v16h aI0 = lds_afrag(arI, hi * 8), aI1 = lds_afrag(arI, 32 + hi * 8);
      v16h aH0 = lds_afrag(arH, hi * 8), aH1 = lds_afrag(arH, 32 + hi * 8);
      v8f ci, ch;
#pragma unroll
      for (int r = 0; r < 8; ++r) { ci[r] = bv2i; ch[r] = bv2h; }
      ci = wmma16(aI0, B2i0, ci); ci = wmma16(aI1, B2i1, ci);
      ch = wmma16(aH0, B2h0, ch); ch = wmma16(aH1, B2h1, ch);
      const int rb = m0 + hi * 8;
#pragma unroll
      for (int r = 0; r < 8; ++r) {
        GIs[(rb + r) * GSTRIDE + col] = ci[r];
        GHs[(rb + r) * GSTRIDE + col] = ch[r];
      }
    }
    __syncthreads();
    for (int i = tid; i < BATCH * HID; i += 384) {
      const int b = i >> 6, j = i & 63;
      const float* ib = GIs + b * GSTRIDE;
      const float* hb = GHs + b * GSTRIDE;
      float r = sigm(ib[j] + hb[j]);
      float z = sigm(ib[64 + j] + hb[64 + j]);
      float n = tanh_f(ib[128 + j] + r * hb[128 + j]);
      float hp = (float)h2s[i];
      float hn = (1.f - z) * n + z * hp;
      h2s[i] = (_Float16)hn;
      out[(size_t)b * (T_STEPS * HID) + (size_t)t * HID + j] = hn;
    }
    __syncthreads();
  }
}

// ---------------------------------------------------------------- launcher
extern "C" void kernel_launch(void* const* d_in, const int* in_sizes, int n_in,
                              void* d_out, int out_size, void* d_ws, size_t ws_size,
                              hipStream_t stream) {
  const float* x    = (const float*)d_in[0];
  const float* W0ih = (const float*)d_in[1];
  const float* W0hh = (const float*)d_in[2];
  const float* b0ih = (const float*)d_in[3];
  const float* b0hh = (const float*)d_in[4];
  const float* W1ih = (const float*)d_in[5];
  const float* W1hh = (const float*)d_in[6];
  const float* b1ih = (const float*)d_in[7];
  const float* b1hh = (const float*)d_in[8];
  const float* W2ih = (const float*)d_in[9];
  const float* W2hh = (const float*)d_in[10];
  const float* b2ih = (const float*)d_in[11];
  const float* b2hh = (const float*)d_in[12];
  float* out = (float*)d_out;

  // workspace layout (all offsets 16B aligned); needs ~50.7 MB
  char* ws = (char*)d_ws;
  const size_t GI0_BYTES = (size_t)T_STEPS * BATCH * G3 * sizeof(float); // 50331648
  float*    gi0    = (float*)ws;
  int*      flags  = (int*)(ws + GI0_BYTES);                 // 2048 B
  _Float16* w0ih_h = (_Float16*)(ws + GI0_BYTES + 2048);     // 196608 B
  _Float16* w0hh_h = (_Float16*)(ws + GI0_BYTES + 2048 + 196608);
  _Float16* w1ih_h = (_Float16*)(ws + GI0_BYTES + 2048 + 196608 + 24576);
  _Float16* w1hh_h = (_Float16*)(ws + GI0_BYTES + 2048 + 196608 + 2 * 24576);
  _Float16* w2ih_h = (_Float16*)(ws + GI0_BYTES + 2048 + 196608 + 3 * 24576);
  _Float16* w2hh_h = (_Float16*)(ws + GI0_BYTES + 2048 + 196608 + 4 * 24576);

  gru_prep_k<<<96, 256, 0, stream>>>(W0ih, W0hh, W1ih, W1hh, W2ih, W2hh,
                                     w0ih_h, w0hh_h, w1ih_h, w1hh_h, w2ih_h, w2hh_h);
  gru_flags_k<<<T_STEPS, 256, 0, stream>>>(x, flags);
  // M = 65536 -> 4096 m-tiles, 8 waves/block -> 512 blocks
  gru_gemm_x_k<<<512, 256, 0, stream>>>(x, w0ih_h, b0ih, gi0);
  gru_rec_k<<<1, 384, 0, stream>>>(gi0, flags, w0hh_h, w1ih_h, w1hh_h, w2ih_h, w2hh_h,
                                   b0hh, b1ih, b1hh, b2ih, b2hh, out);
}
